// GAT_26757646254163
// MI455X (gfx1250) — compile-verified, matrix operand order checked by reading, not run on hardware
//
#include <hip/hip_runtime.h>
#include <hip/hip_bf16.h>
#include <cstdint>
#include <cstddef>

// ---------------------------------------------------------------------------
// GAT forward for MI455X (gfx1250, wave32, WMMA).
//  - GEMMs: bf16 A/B, f32 accumulate via v_wmma_f32_16x16x32_bf16,
//    double-buffered LDS staging, branch-free global_load_async_to_lds_b128
//    (A rows padded to 128, B pre-transposed & padded) with sync fallback.
//  - Edge softmax: 3-pass (atomic-max key, atomic-add sum, weighted scatter)
//  - Node epilogue: ELU + LayerNorm + leaky + residual, one wave per node
// ---------------------------------------------------------------------------

typedef __attribute__((ext_vector_type(16))) __bf16 v16bf;
typedef __attribute__((ext_vector_type(8)))  float  v8f;

union FragBF { uint4 u[2]; v16bf v; };

#define DEV __device__ __forceinline__

#if defined(__has_builtin)
#if __has_builtin(__builtin_amdgcn_global_load_async_to_lds_b128) && \
    __has_builtin(__builtin_amdgcn_s_wait_asynccnt)
#define GAT_ASYNC_LDS 1
#endif
#endif

// Builtin signature (probe-confirmed): (v4i addrspace(1)*, v4i addrspace(3)*,
// imm offset, imm cpol).
typedef int vint4 __attribute__((vector_size(16)));
typedef __attribute__((address_space(1))) vint4 as1_vint4;
typedef __attribute__((address_space(3))) vint4 as3_vint4;

DEV void gat_async_wait() {
#ifdef GAT_ASYNC_LDS
    __builtin_amdgcn_s_wait_asynccnt(0);
#endif
}

DEV void gat_copy16(const uint4* gp, uint4* lp) {
#ifdef GAT_ASYNC_LDS
    __builtin_amdgcn_global_load_async_to_lds_b128(
        (as1_vint4*)(vint4*)(void*)gp, (as3_vint4*)(vint4*)(void*)lp, 0, 0);
#else
    *lp = *gp;
#endif
}

DEV float leaky02(float x) { return x >= 0.f ? x : 0.2f * x; }

// Monotone float<->int key so atomicMax(int) implements float max.
DEV int   fkey(float x)  { int b = __float_as_int(x); return b >= 0 ? b : (b ^ 0x7FFFFFFF); }
DEV float fdecode(int k) { return __int_as_float(k >= 0 ? k : (k ^ 0x7FFFFFFF)); }

DEV unsigned short f2bf(float x) {
    unsigned u = __float_as_uint(x);
    unsigned r = (u >> 16) & 1u;                 // round-to-nearest-even
    return (unsigned short)((u + 0x7FFFu + r) >> 16);
}

// ---------------------------------------------------------------------------
// f32 -> bf16 conversion (grid-stride)
// ---------------------------------------------------------------------------
__global__ void k_cvt_bf16(const float* __restrict__ src,
                           unsigned short* __restrict__ dst, int n) {
    int i = blockIdx.x * blockDim.x + threadIdx.x;
    int stride = gridDim.x * blockDim.x;
    for (; i < n; i += stride) dst[i] = f2bf(src[i]);
}

// Zero u16 range (pads A-matrix rows so GEMM staging is branch-free).
__global__ void k_zero_u16(unsigned short* __restrict__ p, int n) {
    int i = blockIdx.x * 256 + threadIdx.x;
    if (i < n) p[i] = 0;
}

// Wt[n][k] = bf16(W[k][n]); rows n >= Ncols zero-padded up to NP.
// Gives the GEMM a contiguous 64B-per-row B layout for async DMA staging.
__global__ void k_cvt_transpose(const float* __restrict__ W,
                                unsigned short* __restrict__ Wt,
                                int Ncols, int NP) {
    int idx = blockIdx.x * 256 + threadIdx.x;
    if (idx >= NP * 256) return;
    int n = idx >> 8, k = idx & 255;
    float v = (n < Ncols) ? W[(size_t)k * Ncols + n] : 0.f;
    Wt[idx] = f2bf(v);
}

// ---------------------------------------------------------------------------
// C[M x Ncols] = A[Mpad x 256](bf16) * Wt[NP x 256](bf16, pre-transposed), f32.
// Block: 256 threads = 8 waves. Block tile 128(M) x 32(N); each wave owns a
// 16x32 slice (two WMMA accumulators sharing one A fragment). K staged via
// double-buffered LDS in steps of 32, all movement via async-to-LDS DMA.
// A is padded to a multiple of 128 rows, Wt to a multiple of 32 rows, so the
// staging loops carry no bounds checks (no exec-mask churn in the hot loop).
// Fragment layout per ISA 7.12.2 (16-bit A 16x32): lanes 0-15 hold K 0..7 &
// 16..23, lanes 16-31 hold K 8..15 & 24..31; B symmetric (lane = column).
// ---------------------------------------------------------------------------
__global__ void k_gemm_bf16_wmma(const unsigned short* __restrict__ A,
                                 const unsigned short* __restrict__ Bt,
                                 float* __restrict__ C, int M, int Ncols) {
    __shared__ uint4 sA[2][512];    // 128 x 32 bf16 per buffer (64B rows)
    __shared__ uint4 sB4[2][128];   // 32(n) x 32(k) bf16 per buffer

    const int t    = threadIdx.x;
    const int wave = t >> 5;
    const int lane = t & 31;
    const int rowBase = blockIdx.x * 128;
    const int nBase   = blockIdx.y * 32;
    const int K = 256;

    const int l16 = lane & 15;   // A row within tile / B column within group
    const int kg  = lane >> 4;   // K-group select

    v8f acc0, acc1;
#pragma unroll
    for (int i = 0; i < 8; ++i) { acc0[i] = 0.f; acc1[i] = 0.f; }

    auto stage = [&](int buf, int k0) {
        // A tile: 128x32 bf16 = 512 uint4, 2 per thread, branch-free.
#pragma unroll
        for (int it = 0; it < 2; ++it) {
            int idx = t + it * 256;
            int r = idx >> 2, cc = idx & 3;
            const uint4* gp = (const uint4*)(A + (size_t)(rowBase + r) * K + k0) + cc;
            gat_copy16(gp, &sA[buf][idx]);
        }
        // B tile: 32x32 bf16 = 128 uint4, contiguous rows of Wt, branch-free.
        if (t < 128) {
            int n = t >> 2, cc = t & 3;
            const uint4* gp = (const uint4*)(Bt + (size_t)(nBase + n) * K + k0) + cc;
            gat_copy16(gp, &sB4[buf][t]);
        }
    };

    int buf = 0;
    stage(0, 0);
    gat_async_wait();
    __syncthreads();

    for (int k0 = 0; k0 < K; k0 += 32) {
        int nk = k0 + 32;
        if (nk < K) stage(buf ^ 1, nk);       // prefetch next K-tile

        FragBF a, b0, b1;
        const uint4* rowA = sA[buf] + (wave * 16 + l16) * 4;
        a.u[0] = rowA[kg];                    // K = kg*8 .. kg*8+7
        a.u[1] = rowA[2 + kg];                // K = 16+kg*8 ..
        const uint4* rB0 = sB4[buf] + l16 * 4;
        b0.u[0] = rB0[kg];
        b0.u[1] = rB0[2 + kg];
        const uint4* rB1 = sB4[buf] + (16 + l16) * 4;
        b1.u[0] = rB1[kg];
        b1.u[1] = rB1[2 + kg];

        acc0 = __builtin_amdgcn_wmma_f32_16x16x32_bf16(
                   false, a.v, false, b0.v, (short)0, acc0, false, false);
        acc1 = __builtin_amdgcn_wmma_f32_16x16x32_bf16(
                   false, a.v, false, b1.v, (short)0, acc1, false, false);

        gat_async_wait();                     // next-buf DMA landed
        __syncthreads();                      // everyone done reading buf
        buf ^= 1;
    }

    // C/D layout: lane L holds col N=L&15, rows M = v + 8*(L>>4)
#pragma unroll
    for (int v = 0; v < 8; ++v) {
        int grow = rowBase + wave * 16 + v + 8 * kg;
        if (grow < M) {
            int c0 = nBase + l16;
            int c1 = nBase + 16 + l16;
            if (c0 < Ncols) C[(size_t)grow * Ncols + c0] = acc0[v];
            if (c1 < Ncols) C[(size_t)grow * Ncols + c1] = acc1[v];
        }
    }
}

// ---------------------------------------------------------------------------
// Hidden-layer node prep: el/er = <feat[n,h,:], al/ar[h,:]>, init m/s, zero rst.
// One wave per node (8 nodes per 256-thread block), 8 features per lane.
// ---------------------------------------------------------------------------
__global__ void k_node_prep_h(const float* __restrict__ feat,
                              const float* __restrict__ al,
                              const float* __restrict__ ar,
                              float* __restrict__ el, float* __restrict__ er,
                              int* __restrict__ mkey, float* __restrict__ ssum,
                              float* __restrict__ rst, int N) {
    int wave = threadIdx.x >> 5, lane = threadIdx.x & 31;
    int n = blockIdx.x * 8 + wave;
    if (n >= N) return;
    const float* fr = feat + (size_t)n * 256 + lane * 8;
    float4 f0 = *(const float4*)(fr);
    float4 f1 = *(const float4*)(fr + 4);
    float fv[8] = {f0.x, f0.y, f0.z, f0.w, f1.x, f1.y, f1.z, f1.w};
    const float* alp = al + lane * 8;
    const float* arp = ar + lane * 8;
    float pl = 0.f, pr = 0.f;
#pragma unroll
    for (int i = 0; i < 8; ++i) { pl += fv[i] * alp[i]; pr += fv[i] * arp[i]; }
#pragma unroll
    for (int o = 4; o > 0; o >>= 1) {
        pl += __shfl_xor(pl, o, 8);
        pr += __shfl_xor(pr, o, 8);
    }
    if ((lane & 7) == 0) {
        int h = lane >> 3;
        el[n * 4 + h] = pl;  er[n * 4 + h] = pr;
        mkey[n * 4 + h] = (int)0x80000000;  ssum[n * 4 + h] = 0.f;
    }
    float4 z = make_float4(0.f, 0.f, 0.f, 0.f);
    *(float4*)(rst + (size_t)n * 256 + lane * 8)     = z;
    *(float4*)(rst + (size_t)n * 256 + lane * 8 + 4) = z;
}

// Output-layer node prep (H=1, D=40).
__global__ void k_node_prep_o(const float* __restrict__ feat,
                              const float* __restrict__ al,
                              const float* __restrict__ ar,
                              float* __restrict__ el, float* __restrict__ er,
                              int* __restrict__ mkey, float* __restrict__ ssum,
                              float* __restrict__ rst, int N) {
    int wave = threadIdx.x >> 5, lane = threadIdx.x & 31;
    int n = blockIdx.x * 8 + wave;
    if (n >= N) return;
    float pl = 0.f, pr = 0.f;
#pragma unroll
    for (int it = 0; it < 2; ++it) {
        int d = lane + it * 32;
        if (d < 40) {
            float f = feat[(size_t)n * 40 + d];
            pl += f * al[d];  pr += f * ar[d];
            rst[(size_t)n * 40 + d] = 0.f;
        }
    }
#pragma unroll
    for (int o = 16; o > 0; o >>= 1) {
        pl += __shfl_xor(pl, o, 32);
        pr += __shfl_xor(pr, o, 32);
    }
    if (lane == 0) {
        el[n] = pl;  er[n] = pr;
        mkey[n] = (int)0x80000000;  ssum[n] = 0.f;
    }
}

// ---------------------------------------------------------------------------
// Edge pass 1: per-destination running max of leaky(el[src]+er[dst]).
// ---------------------------------------------------------------------------
template <int H>
__global__ void k_edge_max(const float* __restrict__ el,
                           const float* __restrict__ er,
                           int* __restrict__ mkey,
                           const int* __restrict__ src,
                           const int* __restrict__ dst, int E) {
    int i = blockIdx.x * blockDim.x + threadIdx.x;
    if (i >= E) return;
    int s = src[i], d = dst[i];
#pragma unroll
    for (int h = 0; h < H; ++h) {
        float e = leaky02(el[s * H + h] + er[d * H + h]);
        atomicMax(&mkey[d * H + h], fkey(e));
    }
}

// Edge pass 2: partition function s[dst] += exp(e - m[dst]).
template <int H>
__global__ void k_edge_sum(const float* __restrict__ el,
                           const float* __restrict__ er,
                           const int* __restrict__ mkey,
                           float* __restrict__ ssum,
                           const int* __restrict__ src,
                           const int* __restrict__ dst, int E) {
    int i = blockIdx.x * blockDim.x + threadIdx.x;
    if (i >= E) return;
    int s = src[i], d = dst[i];
#pragma unroll
    for (int h = 0; h < H; ++h) {
        float e = leaky02(el[s * H + h] + er[d * H + h]);
        atomicAdd(&ssum[d * H + h], __expf(e - fdecode(mkey[d * H + h])));
    }
}

// Edge pass 3: rst[dst,f..f+3] += alpha * feat[src,f..f+3].
// One thread per (edge, feature-quad); float4 gather + 4 f32 L2 atomics.
// feat (102 MB) is L2-resident on MI455X (192 MB L2) -> gather at L2 rate.
template <int F, int Dd>
__global__ void k_edge_aggr(const float* __restrict__ feat,
                            const float* __restrict__ el,
                            const float* __restrict__ er,
                            const int* __restrict__ mkey,
                            const float* __restrict__ ssum,
                            float* __restrict__ rst,
                            const int* __restrict__ src,
                            const int* __restrict__ dst, int E) {
    constexpr int H  = F / Dd;
    constexpr int FV = F / 4;                  // feature quads per edge
    long long idx   = (long long)blockIdx.x * 256 + threadIdx.x;
    long long total = (long long)E * FV;
    if (idx >= total) return;
    int e = (int)(idx / FV);
    int f = 4 * (int)(idx - (long long)e * FV);
    int h = f / Dd;                            // uniform over the quad
    int s = src[e], d = dst[e];
    const float* fp = feat + (size_t)s * F + f;
    __builtin_prefetch(fp, 0, 3);              // global_prefetch_b8
    float lg    = leaky02(el[s * H + h] + er[d * H + h]);
    float alpha = __expf(lg - fdecode(mkey[d * H + h])) / ssum[d * H + h];
    float4 fv = *(const float4*)fp;
    float* rp = rst + (size_t)d * F + f;
    atomicAdd(rp + 0, alpha * fv.x);
    atomicAdd(rp + 1, alpha * fv.y);
    atomicAdd(rp + 2, alpha * fv.z);
    atomicAdd(rp + 3, alpha * fv.w);
}

// ---------------------------------------------------------------------------
// Hidden epilogue: +bias -> ELU -> LayerNorm -> leaky(0.2) -> +residual.
// One wave per node; mean/var via full-wave shfl reduction (wave32).
// Writes f32 h and bf16 h (next GEMM's A matrix).
// ---------------------------------------------------------------------------
__global__ void k_epilogue(const float* __restrict__ rst,
                           const float* __restrict__ bias,
                           const float* __restrict__ g,
                           const float* __restrict__ b,
                           const float* __restrict__ hres,
                           float* __restrict__ hout,
                           unsigned short* __restrict__ hbf, int N) {
    int wave = threadIdx.x >> 5, lane = threadIdx.x & 31;
    int n = blockIdx.x * 8 + wave;
    if (n >= N) return;
    const float* rp = rst + (size_t)n * 256 + lane * 8;
    float4 a0 = *(const float4*)rp;
    float4 a1 = *(const float4*)(rp + 4);
    float v[8] = {a0.x, a0.y, a0.z, a0.w, a1.x, a1.y, a1.z, a1.w};
    float s1 = 0.f, s2 = 0.f;
#pragma unroll
    for (int i = 0; i < 8; ++i) {
        float x = v[i] + bias[lane * 8 + i];
        x = x > 0.f ? x : (__expf(x) - 1.f);        // ELU
        v[i] = x;  s1 += x;  s2 += x * x;
    }
#pragma unroll
    for (int o = 16; o > 0; o >>= 1) {
        s1 += __shfl_xor(s1, o, 32);
        s2 += __shfl_xor(s2, o, 32);
    }
    float mu   = s1 * (1.f / 256.f);
    float var  = s2 * (1.f / 256.f) - mu * mu;
    float rstd = rsqrtf(var + 1e-5f);
#pragma unroll
    for (int i = 0; i < 8; ++i) {
        int f = lane * 8 + i;
        float y = (v[i] - mu) * rstd * g[f] + b[f];
        y = leaky02(y);
        y += hres[(size_t)n * 256 + f];             // residual (256 == 256)
        hout[(size_t)n * 256 + f] = y;
        hbf [(size_t)n * 256 + f] = f2bf(y);
    }
}

// Output epilogue: logits = rst + bias_o (H=1, sum over heads is identity).
__global__ void k_out_final(const float* __restrict__ rst,
                            const float* __restrict__ bias,
                            float* __restrict__ out, int N) {
    int i = blockIdx.x * 256 + threadIdx.x;
    if (i >= N * 40) return;
    out[i] = rst[i] + bias[i % 40];
}

// ---------------------------------------------------------------------------
// Host orchestration
// ---------------------------------------------------------------------------
extern "C" void kernel_launch(void* const* d_in, const int* in_sizes, int n_in,
                              void* d_out, int out_size, void* d_ws, size_t ws_size,
                              hipStream_t stream) {
    (void)n_in; (void)out_size; (void)ws_size;
    const float* x      = (const float*)d_in[0];
    const float* W_h    = (const float*)d_in[1];
    const float* al_h   = (const float*)d_in[2];
    const float* ar_h   = (const float*)d_in[3];
    const float* bias_h = (const float*)d_in[4];
    const float* ln_g   = (const float*)d_in[5];
    const float* ln_b   = (const float*)d_in[6];
    const float* W_o    = (const float*)d_in[7];
    const float* al_o   = (const float*)d_in[8];
    const float* ar_o   = (const float*)d_in[9];
    const float* bias_o = (const float*)d_in[10];
    const int*   esrc   = (const int*)d_in[11];
    const int*   edst   = (const int*)d_in[12];

    const int N    = in_sizes[0] / 256;
    const int E    = in_sizes[11];
    const int Npad = (N + 127) & ~127;         // branch-free GEMM staging

    char*  ws  = (char*)d_ws;
    size_t off = 0;
    auto alloc = [&](size_t bytes) -> void* {
        void* p = (void*)(ws + off);
        off += (bytes + 255) & ~(size_t)255;
        return p;
    };
    float*          feat = (float*)alloc((size_t)N * 256 * 4);
    float*          rst  = (float*)alloc((size_t)N * 256 * 4);
    float*          hbuf = (float*)alloc((size_t)N * 256 * 4);
    unsigned short* hbf  = (unsigned short*)alloc((size_t)Npad * 256 * 2);
    unsigned short* Wt   = (unsigned short*)alloc((size_t)256 * 256 * 2);
    float*          el   = (float*)alloc((size_t)N * 4 * 4);
    float*          er   = (float*)alloc((size_t)N * 4 * 4);
    int*            mkey = (int*)alloc((size_t)N * 4 * 4);
    float*          ssum = (float*)alloc((size_t)N * 4 * 4);

    dim3 blk(256);
    const int eBlocks = (E + 255) / 256;
    const int nBlocks = (N + 7) / 8;

    // Zero A-matrix pad rows once per call (epilogues never touch them).
    if (Npad > N) {
        int padElems = (Npad - N) * 256;
        k_zero_u16<<<(padElems + 255) / 256, blk, 0, stream>>>(hbf + (size_t)N * 256, padElems);
    }
    // Layer 0 A-matrix: bf16 copy of x.
    k_cvt_bf16<<<1024, blk, 0, stream>>>(x, hbf, N * 256);

    for (int l = 0; l < 3; ++l) {
        const float* hin = (l == 0) ? x : hbuf;
        k_cvt_transpose<<<(256 * 256 + 255) / 256, blk, 0, stream>>>(
            W_h + (size_t)l * 256 * 256, Wt, 256, 256);

        dim3 ggrid(Npad / 128, 8);            // 8 x 32 = 256 output cols
        k_gemm_bf16_wmma<<<ggrid, blk, 0, stream>>>(hbf, Wt, feat, N, 256);

        k_node_prep_h<<<nBlocks, blk, 0, stream>>>(feat, al_h + l * 256, ar_h + l * 256,
                                                   el, er, mkey, ssum, rst, N);
        k_edge_max<4><<<eBlocks, blk, 0, stream>>>(el, er, mkey, esrc, edst, E);
        k_edge_sum<4><<<eBlocks, blk, 0, stream>>>(el, er, mkey, ssum, esrc, edst, E);

        long long tot = (long long)E * (256 / 4);
        k_edge_aggr<256, 64><<<(unsigned)((tot + 255) / 256), blk, 0, stream>>>(
            feat, el, er, mkey, ssum, rst, esrc, edst, E);

        k_epilogue<<<nBlocks, blk, 0, stream>>>(rst, bias_h + l * 256, ln_g + l * 256,
                                                ln_b + l * 256, hin, hbuf, hbf, N);
    }

    // Output GAT layer (H=1, D=40, no activation).
    k_cvt_transpose<<<(64 * 256 + 255) / 256, blk, 0, stream>>>(W_o, Wt, 40, 64);
    dim3 ogrid(Npad / 128, 2);                // 2 x 32 covers 40 cols (padded)
    k_gemm_bf16_wmma<<<ogrid, blk, 0, stream>>>(hbf, Wt, feat, N, 40);

    k_node_prep_o<<<nBlocks, blk, 0, stream>>>(feat, al_o, ar_o, el, er, mkey, ssum, rst, N);
    k_edge_max<1><<<eBlocks, blk, 0, stream>>>(el, er, mkey, esrc, edst, E);
    k_edge_sum<1><<<eBlocks, blk, 0, stream>>>(el, er, mkey, ssum, esrc, edst, E);

    long long tot_o = (long long)E * (40 / 4);
    k_edge_aggr<40, 40><<<(unsigned)((tot_o + 255) / 256), blk, 0, stream>>>(
        feat, el, er, mkey, ssum, rst, esrc, edst, E);

    k_out_final<<<(N * 40 + 255) / 256, blk, 0, stream>>>(rst, bias_o, (float*)d_out, N);
}
